// GeoNGNN_67534065762911
// MI455X (gfx1250) — matrix operand order repaired and checked by reading, not compiled, not run on hardware
//
#include <hip/hip_runtime.h>
#include <math.h>

#define HID 128
#define EF_ 32

typedef __attribute__((ext_vector_type(16))) _Float16 v16h;
typedef __attribute__((ext_vector_type(4)))  _Float16 v4h;
typedef __attribute__((ext_vector_type(8)))  float    v8f;
typedef __attribute__((ext_vector_type(4)))  unsigned int tdm_u32x4;
typedef __attribute__((ext_vector_type(8)))  int          tdm_i32x8;
typedef __attribute__((ext_vector_type(4)))  int          tdm_i32x4;

#if defined(__has_builtin)
#if __has_builtin(__builtin_amdgcn_tensor_load_to_lds) && __has_builtin(__builtin_amdgcn_s_wait_tensorcnt)
#define USE_TDM 1
#endif
#endif
#ifndef USE_TDM
#define USE_TDM 0
#endif

__device__ __forceinline__ float silu_f(float x) { return x / (1.f + __expf(-x)); }

__device__ __forceinline__ float cutoff_f(float d, float rc) {
  float dm = fminf(d, rc);
  float v = 0.5f * (__cosf(3.14159265358979323846f * dm / rc) + 1.f);
  return (d < rc) ? v : 0.f;
}

#if USE_TDM
// Issue one TDM 2D tile load: Wt[128][K] f16 (row stride K elems) -> LDS.
// D# per CDNA5 ISA: data_size=1(2B), tile K x 128,
// type=2, no padding/iterate/multicast. 6-arg builtin variant (clang-23 shape).
__device__ __forceinline__ void tdm_load_weights(const _Float16* Wt,
                                                 unsigned lds_addr, int K) {
  unsigned long long ga = (unsigned long long)(uintptr_t)Wt;
  tdm_u32x4 g0;
  g0[0] = 1u;                                   // count=1 (valid descriptor)
  g0[1] = lds_addr;                             // lds_addr bytes
  g0[2] = (unsigned)ga;                         // global_addr[31:0]
  g0[3] = ((unsigned)(ga >> 32) & 0x01FFFFFFu)  // global_addr[56:32]
          | (2u << 30);                         // type=2 (image)
  tdm_i32x8 g1;
  g1[0] = 0x00010000;                           // data_size=1 -> 2 bytes
  g1[1] = (int)((unsigned)K << 16);             // tensor_dim0[15:0]=K
  g1[2] = (int)(128u << 16);                    // tensor_dim1[15:0]=128
  g1[3] = (int)((unsigned)K << 16);             // tile_dim0=K
  g1[4] = 128;                                  // tile_dim1=128, tile_dim2=0
  g1[5] = K;                                    // tensor_dim0_stride[31:0]=K
  g1[6] = 0;
  g1[7] = 0;
  tdm_i32x4 g2 = {0, 0, 0, 0};                  // group 2 unused (<=2D tensor)
  tdm_i32x4 g3 = {0, 0, 0, 0};                  // group 3 unused
  tdm_i32x8 g4 = {0, 0, 0, 0, 0, 0, 0, 0};      // extra group (6-arg variant)
  __builtin_amdgcn_tensor_load_to_lds(g0, g1, g2, g3, g4, 0);
}
#endif

// ============================================================================
// WMMA GEMM:  Y[M,128] = epi( X[M,K] @ W[K,128] + bias ),  K compile-time.
//   Weights pre-converted to f16, N-major: Wt[n*K + k]; whole B operand is
//   pulled into LDS once (TDM tensor_load_to_lds when available).
//   X = [XA | XB] along K when XB != nullptr (each [M,128]).
//   M must be a multiple of 128 (true for every call site).
// 256 threads = 8 waves; each wave computes a 16x128 slab via 8 WMMA tiles.
// ============================================================================
template <int K>
__global__ __launch_bounds__(256) void gemm128_kernel(
    const float* __restrict__ XA, const float* __restrict__ XB,
    const _Float16* __restrict__ Wt, const float* __restrict__ bias,
    const float* __restrict__ mul, const float* __restrict__ rowscale,
    const float* __restrict__ res1, const float* __restrict__ res2,
    float* __restrict__ Y, int act)
{
  __shared__ _Float16 sX[128 * 32];   // [row][k-chunk]  f16
  __shared__ _Float16 sW[128 * K];    // [n][k]          f16 (full B operand)
  const int tid  = threadIdx.x;
  const int lane = tid & 31;
  const int wave = tid >> 5;
  const int row0 = blockIdx.x * 128;

  // ---- load full weight tile into LDS ----
#if USE_TDM
  if (tid == 0) tdm_load_weights(Wt, (unsigned)(uintptr_t)&sW[0], K);
  __builtin_amdgcn_s_wait_tensorcnt((short)0);  // TENSORcnt==0 waves pass through
#else
  {
    const uint4* s = (const uint4*)Wt;
    uint4* d = (uint4*)sW;
    for (int i = tid; i < (128 * K) / 8; i += 256) d[i] = s[i];
  }
#endif

  v8f acc[8];
#pragma unroll
  for (int t = 0; t < 8; ++t)
#pragma unroll
    for (int j = 0; j < 8; ++j) acc[t][j] = 0.f;

  constexpr int NK = K / 32;
#pragma unroll
  for (int kc = 0; kc < NK; ++kc) {
    const int kbase = kc * 32;
    // per-chunk X source (resolves the [XA|XB] concat at scalar level)
    const float* Xc;
    int ld;
    if (K == 256) {
      Xc = (kbase < HID) ? (XA + kbase) : (XB + (kbase - HID));
      ld = HID;
    } else {
      Xc = XA + kbase;
      ld = K;
    }
    __syncthreads();   // also orders first chunk after the weight-tile load
    // Stage X chunk: 128 rows x 32 cols, float4 loads -> packed f16 stores
#pragma unroll
    for (int i = 0; i < 4; ++i) {
      int idx = i * 256 + tid;          // 1024 float4 groups
      int r = idx >> 3, q = idx & 7;
      float4 v = *(const float4*)&Xc[(size_t)(row0 + r) * ld + q * 4];
      v4h h;
      h[0] = (_Float16)v.x; h[1] = (_Float16)v.y;
      h[2] = (_Float16)v.z; h[3] = (_Float16)v.w;
      *(v4h*)&sX[r * 32 + q * 4] = h;
    }
    if (K > 32 && kc + 1 < NK && !XB) {  // prefetch next chunk (global_prefetch)
      __builtin_prefetch(
          &XA[(size_t)(row0 + (tid & 127)) * K + kbase + 32 + (tid >> 7) * 16], 0, 3);
    }
    __syncthreads();

    const int m = lane & 15;
    // A fragment: contiguous k order, same as B (lanes 0-15: K0-15; 16-31: K16-31)
    v16h a;
    {
      const _Float16* xr = &sX[(wave * 16 + m) * 32];
      const int kb = (lane < 16) ? 0 : 16;
#pragma unroll
      for (int i = 0; i < 16; ++i) { a[i] = xr[kb + i]; }
    }
#pragma unroll
    for (int nt = 0; nt < 8; ++nt) {
      // B fragment: n = nt*16 + (lane&15); lanes 0-15 K0-15, lanes 16-31 K16-31
      v16h b;
      const _Float16* wc = &sW[(size_t)(nt * 16 + m) * K + kbase + ((lane < 16) ? 0 : 16)];
#pragma unroll
      for (int i = 0; i < 16; ++i) b[i] = wc[i];
      acc[nt] = __builtin_amdgcn_wmma_f32_16x16x32_f16(
          false, a, false, b, (short)0, acc[nt], false, false);
    }
  }

  // Epilogue: C/D layout -> lane<16: rows 0-7, lane>=16: rows 8-15; col = lane&15
  const int rbase = row0 + wave * 16 + ((lane < 16) ? 0 : 8);
  const int cbase = lane & 15;
#pragma unroll
  for (int nt = 0; nt < 8; ++nt) {
    const int col = nt * 16 + cbase;
    const float bc = bias ? bias[col] : 0.f;
#pragma unroll
    for (int j = 0; j < 8; ++j) {
      const int row = rbase + j;
      float v = acc[nt][j] + bc;
      if (act) v = silu_f(v);
      if (rowscale) v *= rowscale[row];
      const size_t o = (size_t)row * HID + col;
      if (mul)  v *= mul[o];
      if (res1) v += res1[o];
      if (res2) v += res2[o];
      Y[o] = v;
    }
  }
}

// ---- weight convert+transpose: Wt[n*K+k] = (f16) W[k*128+n]
__global__ void wcvt_kernel(const float* __restrict__ W,
                            _Float16* __restrict__ Wt, int K)
{
  int t = blockIdx.x * blockDim.x + threadIdx.x;
  int k = t >> 7, n = t & 127;
  if (k < K) Wt[(size_t)n * K + k] = (_Float16)W[(size_t)k * HID + n];
}

// ---- embedding gather: dst[i,:] = table[sel,:], sel = idxB ? idxB[idxA[i]] : idxA[i]
__global__ void gather_emb_kernel(float* __restrict__ dst,
                                  const float* __restrict__ table,
                                  const int* __restrict__ idxA,
                                  const int* __restrict__ idxB, int n)
{
  int t = blockIdx.x * blockDim.x + threadIdx.x;
  int i = t >> 7, c = t & 127;
  if (i < n) {
    int sel = idxA[i];
    if (idxB) sel = idxB[sel];
    dst[(size_t)i * HID + c] = table[(size_t)sel * HID + c];
  }
}

// ---- inner node geometry: gather positions, distance-to-center envelope
__global__ void inner_geom_kernel(const float* __restrict__ pos,
                                  const int* __restrict__ nidx,
                                  const int* __restrict__ cidx,
                                  float* __restrict__ ipos,
                                  float* __restrict__ env, int n, float rc)
{
  int i = blockIdx.x * blockDim.x + threadIdx.x;
  if (i < n) {
    int a = nidx[i], b = cidx[i];
    float px = pos[3 * a], py = pos[3 * a + 1], pz = pos[3 * a + 2];
    float cx = pos[3 * b], cy = pos[3 * b + 1], cz = pos[3 * b + 2];
    ipos[3 * i] = px; ipos[3 * i + 1] = py; ipos[3 * i + 2] = pz;
    float dx = px - cx, dy = py - cy, dz = pz - cz;
    env[i] = cutoff_f(sqrtf(dx * dx + dy * dy + dz * dz), rc);
  }
}

// ---- edge geometry: distance -> RBF features + cosine cutoff
__global__ void edge_geom_kernel(const int* __restrict__ ei,
                                 const float* __restrict__ P,
                                 float* __restrict__ ef,
                                 float* __restrict__ conv, int E2, float rc)
{
  int e = blockIdx.x * blockDim.x + threadIdx.x;
  if (e < E2) {
    int s = ei[e], d = ei[E2 + e];
    float dx = P[3 * d] - P[3 * s];
    float dy = P[3 * d + 1] - P[3 * s + 1];
    float dz = P[3 * d + 2] - P[3 * s + 2];
    float dist = sqrtf(dx * dx + dy * dy + dz * dz);
    conv[e] = cutoff_f(dist, rc);
    float step = rc / 31.f;
    float4* row = (float4*)&ef[(size_t)e * EF_];
#pragma unroll
    for (int q = 0; q < 8; ++q) {
      float4 v;
      float t0 = dist - step * (4 * q + 0);
      float t1 = dist - step * (4 * q + 1);
      float t2 = dist - step * (4 * q + 2);
      float t3 = dist - step * (4 * q + 3);
      v.x = __expf(-10.f * t0 * t0); v.y = __expf(-10.f * t1 * t1);
      v.z = __expf(-10.f * t2 * t2); v.w = __expf(-10.f * t3 * t3);
      row[q] = v;
    }
  }
}

// ---- message + scatter-add: agg[dst] += scalar[src] * g   (conv folded in g)
__global__ void msg_scatter_kernel(const float* __restrict__ scalar,
                                   const float* __restrict__ g,
                                   const int* __restrict__ src,
                                   const int* __restrict__ dst,
                                   float* __restrict__ agg, int E2)
{
  int t = blockIdx.x * blockDim.x + threadIdx.x;
  int e = t >> 7, c = t & 127;
  if (e < E2) {
    float v = scalar[(size_t)src[e] * HID + c] * g[(size_t)t];
    unsafeAtomicAdd(&agg[(size_t)dst[e] * HID + c], v);
  }
}

// ---- segment pool: out[seg[i]] += x[i] * (scale ? scale[i] : 1)
__global__ void pool_scatter_kernel(const float* __restrict__ x,
                                    const float* __restrict__ scale,
                                    const int* __restrict__ seg,
                                    float* __restrict__ out, int n)
{
  int t = blockIdx.x * blockDim.x + threadIdx.x;
  int i = t >> 7, c = t & 127;
  if (i < n) {
    float v = x[(size_t)i * HID + c];
    if (scale) v *= scale[i];
    unsafeAtomicAdd(&out[(size_t)seg[i] * HID + c], v);
  }
}

__global__ void fill_kernel(float* __restrict__ p, long n)
{
  long t = (long)blockIdx.x * blockDim.x + threadIdx.x;
  if (t < n) p[t] = 0.f;
}

// ---- head: out[b] = dot(graph[b,:], Wout) * YSTD + YMEAN (one wave32 per b)
__global__ void head_kernel(const float* __restrict__ graph,
                            const float* __restrict__ wout,
                            float* __restrict__ out, float ystd, float ymean)
{
  int b = blockIdx.x, lane = threadIdx.x;
  float s = 0.f;
  for (int i = lane; i < HID; i += 32) s += graph[(size_t)b * HID + i] * wout[i];
#pragma unroll
  for (int off = 16; off > 0; off >>= 1) s += __shfl_down(s, off);
  if (lane == 0) out[b] = s * ystd + ymean;
}

// ============================================================================
extern "C" void kernel_launch(void* const* d_in, const int* in_sizes, int n_in,
                              void* d_out, int out_size, void* d_ws, size_t ws_size,
                              hipStream_t stream)
{
  (void)in_sizes; (void)n_in; (void)out_size; (void)ws_size;
  const int N = 16384, E = 262144, SN = 262144, SE = 524288, B = 256;
  const int L_IN = 3, L_OUT = 4;

  // Inputs in setup_inputs() dict order
  const float* pos          = (const float*)d_in[0];
  const int*   z            = (const int*)d_in[1];
  const int*   edge_index   = (const int*)d_in[2];
  const int*   batch_index  = (const int*)d_in[3];
  const int*   sn_index     = (const int*)d_in[4];
  const int*   sn_center    = (const int*)d_in[5];
  const int*   se_index     = (const int*)d_in[6];
  const int*   sb_index     = (const int*)d_in[7];
  const int*   sn_label     = (const int*)d_in[8];
  const float* inner_z_emb  = (const float*)d_in[9];
  const float* outer_z_emb  = (const float*)d_in[10];
  const float* label_emb    = (const float*)d_in[11];
  const float* label_proj_W = (const float*)d_in[12];
  const float* label_proj_b = (const float*)d_in[13];
  const float* inner_proj_W = (const float*)d_in[14];
  const float* inner_proj_b = (const float*)d_in[15];
  const float* outer_proj_W = (const float*)d_in[16];
  const float* outer_proj_b = (const float*)d_in[17];
  const float* in_Wef       = (const float*)d_in[18];
  const float* in_bef       = (const float*)d_in[19];
  const float* in_Wu        = (const float*)d_in[20];
  const float* in_bu        = (const float*)d_in[21];
  const float* out_Wef      = (const float*)d_in[22];
  const float* out_bef      = (const float*)d_in[23];
  const float* out_Wu       = (const float*)d_in[24];
  const float* out_bu       = (const float*)d_in[25];
  const float* fuse_W       = (const float*)d_in[26];
  const float* fuse_b       = (const float*)d_in[27];
  const float* res_W1       = (const float*)d_in[28];
  const float* res_b1       = (const float*)d_in[29];
  const float* res_W2       = (const float*)d_in[30];
  const float* res_b2       = (const float*)d_in[31];
  const float* output_W     = (const float*)d_in[32];
  float* out = (float*)d_out;

  // ---- workspace layout (f32 region, all sizes multiples of 4 -> 16B align)
  float* w = (float*)d_ws;
  size_t off = 0;
  auto take = [&](size_t n) { float* p = w + off; off += n; return p; };
  float* ipos  = take((size_t)SN * 3);
  float* env   = take((size_t)SN);
  float* efb   = take((size_t)SE * EF_);
  float* convb = take((size_t)SE);
  float* Xa    = take((size_t)SN * HID);
  float* Xb    = take((size_t)SN * HID);
  float* iscal = take((size_t)SN * HID);
  float* gbuf  = take((size_t)SE * HID);
  float* agg   = take((size_t)SN * HID);
  float* senv  = take((size_t)N * HID);
  float* oscal = take((size_t)N * HID);
  float* hbuf  = take((size_t)N * HID);
  float* h1buf = take((size_t)N * HID);
  float* graph = take((size_t)B * HID);
  // ---- f16 weight region (N-major transposed copies)
  _Float16* hws = (_Float16*)(w + off);
  size_t hoff = 0;
  auto takeh = [&](size_t n) { _Float16* p = hws + hoff; hoff += n; return p; };
  _Float16* ipW = takeh(128 * 128);
  _Float16* lpW = takeh(128 * 128);
  _Float16* opW = takeh(128 * 128);
  _Float16* fW  = takeh(128 * 256);
  _Float16* r1W = takeh(128 * 128);
  _Float16* r2W = takeh(128 * 128);
  _Float16* iWef[3]; _Float16* iWu[3];
  for (int l = 0; l < L_IN; ++l) { iWef[l] = takeh(128 * 32); iWu[l] = takeh(128 * 128); }
  _Float16* oWef[4]; _Float16* oWu[4];
  for (int l = 0; l < L_OUT; ++l) { oWef[l] = takeh(128 * 32); oWu[l] = takeh(128 * 128); }

  auto cdiv = [](long a, long b) { return (unsigned)((a + b - 1) / b); };

  // ---- convert all weights to f16 N-major once per call
  wcvt_kernel<<<cdiv(128 * 128, 256), 256, 0, stream>>>(inner_proj_W, ipW, 128);
  wcvt_kernel<<<cdiv(128 * 128, 256), 256, 0, stream>>>(label_proj_W, lpW, 128);
  wcvt_kernel<<<cdiv(128 * 128, 256), 256, 0, stream>>>(outer_proj_W, opW, 128);
  wcvt_kernel<<<cdiv(256 * 128, 256), 256, 0, stream>>>(fuse_W, fW, 256);
  wcvt_kernel<<<cdiv(128 * 128, 256), 256, 0, stream>>>(res_W1, r1W, 128);
  wcvt_kernel<<<cdiv(128 * 128, 256), 256, 0, stream>>>(res_W2, r2W, 128);
  for (int l = 0; l < L_IN; ++l) {
    wcvt_kernel<<<cdiv(32 * 128, 256), 256, 0, stream>>>(in_Wef + (size_t)l * EF_ * HID, iWef[l], 32);
    wcvt_kernel<<<cdiv(128 * 128, 256), 256, 0, stream>>>(in_Wu + (size_t)l * HID * HID, iWu[l], 128);
  }
  for (int l = 0; l < L_OUT; ++l) {
    wcvt_kernel<<<cdiv(32 * 128, 256), 256, 0, stream>>>(out_Wef + (size_t)l * EF_ * HID, oWef[l], 32);
    wcvt_kernel<<<cdiv(128 * 128, 256), 256, 0, stream>>>(out_Wu + (size_t)l * HID * HID, oWu[l], 128);
  }

  // ======================== inner (subgraph) branch =========================
  inner_geom_kernel<<<cdiv(SN, 256), 256, 0, stream>>>(pos, sn_index, sn_center,
                                                       ipos, env, SN, 7.0f);
  gather_emb_kernel<<<cdiv((long)SN * HID, 256), 256, 0, stream>>>(Xa, inner_z_emb, sn_index, z, SN);
  gather_emb_kernel<<<cdiv((long)SN * HID, 256), 256, 0, stream>>>(Xb, label_emb, sn_label, nullptr, SN);
  // Xa = silu(Xa @ inner_proj_W + b)   (in-place: block-private rows)
  gemm128_kernel<128><<<SN / 128, 256, 0, stream>>>(Xa, nullptr, ipW, inner_proj_b,
      nullptr, nullptr, nullptr, nullptr, Xa, 1);
  // iscal = silu(Xb @ label_proj_W + b) * Xa
  gemm128_kernel<128><<<SN / 128, 256, 0, stream>>>(Xb, nullptr, lpW, label_proj_b,
      Xa, nullptr, nullptr, nullptr, iscal, 1);
  edge_geom_kernel<<<cdiv(SE, 256), 256, 0, stream>>>(se_index, ipos, efb, convb, SE, 5.0f);

  for (int l = 0; l < L_IN; ++l) {
    const float* bef = in_bef + (size_t)l * HID;
    const float* bu  = in_bu  + (size_t)l * HID;
    // g = silu(ef @ Wef + bef) * conv[row]
    gemm128_kernel<32><<<SE / 128, 256, 0, stream>>>(efb, nullptr, iWef[l], bef,
        nullptr, convb, nullptr, nullptr, gbuf, 1);
    fill_kernel<<<cdiv((long)SN * HID, 256), 256, 0, stream>>>(agg, (long)SN * HID);
    msg_scatter_kernel<<<cdiv((long)SE * HID, 256), 256, 0, stream>>>(iscal, gbuf,
        se_index, se_index + SE, agg, SE);
    // iscal = iscal + silu(agg @ Wu + bu)
    gemm128_kernel<128><<<SN / 128, 256, 0, stream>>>(agg, nullptr, iWu[l], bu,
        nullptr, nullptr, iscal, nullptr, iscal, 1);
  }
  // subg_env[n] = sum_i (iscal[i] * env[i]) over subg_batch_index
  fill_kernel<<<cdiv((long)N * HID, 256), 256, 0, stream>>>(senv, (long)N * HID);
  pool_scatter_kernel<<<cdiv((long)SN * HID, 256), 256, 0, stream>>>(iscal, env, sb_index, senv, SN);

  // ============================ outer branch ================================
  gather_emb_kernel<<<cdiv((long)N * HID, 256), 256, 0, stream>>>(Xa, outer_z_emb, z, nullptr, N);
  gemm128_kernel<128><<<N / 128, 256, 0, stream>>>(Xa, nullptr, opW, outer_proj_b,
      nullptr, nullptr, nullptr, nullptr, oscal, 1);
  // h = silu([oscal | senv] @ fuse_W + fuse_b)   (K = 256)
  gemm128_kernel<256><<<N / 128, 256, 0, stream>>>(oscal, senv, fW, fuse_b,
      nullptr, nullptr, nullptr, nullptr, hbuf, 1);
  // h1 = silu(h @ res_W1 + b1)
  gemm128_kernel<128><<<N / 128, 256, 0, stream>>>(hbuf, nullptr, r1W, res_b1,
      nullptr, nullptr, nullptr, nullptr, h1buf, 1);
  // oscal = silu(h1 @ res_W2 + b2) + h + oscal
  gemm128_kernel<128><<<N / 128, 256, 0, stream>>>(h1buf, nullptr, r2W, res_b2,
      nullptr, nullptr, hbuf, oscal, oscal, 1);
  edge_geom_kernel<<<cdiv(E, 256), 256, 0, stream>>>(edge_index, pos, efb, convb, E, 10.0f);

  for (int l = 0; l < L_OUT; ++l) {
    const float* bef = out_bef + (size_t)l * HID;
    const float* bu  = out_bu  + (size_t)l * HID;
    gemm128_kernel<32><<<E / 128, 256, 0, stream>>>(efb, nullptr, oWef[l], bef,
        nullptr, convb, nullptr, nullptr, gbuf, 1);
    fill_kernel<<<cdiv((long)N * HID, 256), 256, 0, stream>>>(agg, (long)N * HID);
    msg_scatter_kernel<<<cdiv((long)E * HID, 256), 256, 0, stream>>>(oscal, gbuf,
        edge_index, edge_index + E, agg, E);
    gemm128_kernel<128><<<N / 128, 256, 0, stream>>>(agg, nullptr, oWu[l], bu,
        nullptr, nullptr, oscal, nullptr, oscal, 1);
  }

  // ====================== graph pooling + linear head =======================
  fill_kernel<<<cdiv((long)B * HID, 256), 256, 0, stream>>>(graph, (long)B * HID);
  pool_scatter_kernel<<<cdiv((long)N * HID, 256), 256, 0, stream>>>(oscal, nullptr, batch_index, graph, N);
  head_kernel<<<B, 32, 0, stream>>>(graph, output_W, out, 1.0f, 0.0f);
}
